// InfoNCECauchy_23106924053044
// MI455X (gfx1250) — matrix operand order repaired and checked by reading, not compiled, not run on hardware
//
#include <hip/hip_runtime.h>
#include <hip/hip_bf16.h>

typedef __attribute__((ext_vector_type(16))) __bf16 v16bf;
typedef __attribute__((ext_vector_type(8)))  float  v8f;

#define KDIM 128
#define NROWS 4096   // rows per half (a or b)

__device__ inline v8f wmma_bf16(v16bf a, v16bf b, v8f c) {
    // D = A*B + C, 16x16x32 bf16 -> f32 accum
    return __builtin_amdgcn_wmma_f32_16x16x32_bf16(
        /*neg_a=*/false, a, /*neg_b=*/false, b,
        /*c_mod=*/(short)0, c, /*reuse_a=*/false, /*reuse_b=*/false);
}

union FragU {
    uint4  u[2];
    v16bf  v;
};

// ---------------------------------------------------------------------------
// Kernel 1: split fp32 -> bf16 hi/lo planes, compute fp32 row norms
// ---------------------------------------------------------------------------
__global__ __launch_bounds__(128)
void cvt_norm_kernel(const float* __restrict__ F,
                     __hip_bfloat16* __restrict__ Hi,
                     __hip_bfloat16* __restrict__ Lo,
                     float* __restrict__ norms) {
    __shared__ float sm[128];
    const int row = blockIdx.x;
    const int k   = threadIdx.x;
    const int idx = row * KDIM + k;
    float x = F[idx];
    __hip_bfloat16 h = __float2bfloat16(x);
    float hf = __bfloat162float(h);
    __hip_bfloat16 l = __float2bfloat16(x - hf);
    Hi[idx] = h;
    Lo[idx] = l;
    sm[k] = x * x;
    __syncthreads();
    #pragma unroll
    for (int st = 64; st > 0; st >>= 1) {
        if (k < st) sm[k] += sm[k + st];
        __syncthreads();
    }
    if (k == 0) norms[row] = sm[0];
}

// ---------------------------------------------------------------------------
// Kernel 2: fused Gram + Cauchy + row-reduction.
// Jobs: 0 = aa (masked diag), 1 = bb (masked diag), 2 = ab (rowsum + diag log),
//       3 = ba (rowsum == colsum of ab).
// Wave = 16-row stripe x 1024-column split. Partials: [job][split][4096].
// ---------------------------------------------------------------------------
__global__ __launch_bounds__(256)
void gram_kernel(const __hip_bfloat16* __restrict__ Hi,
                 const __hip_bfloat16* __restrict__ Lo,
                 const float* __restrict__ norms,
                 float* __restrict__ partial,
                 float* __restrict__ diaglog) {
    const int lane = threadIdx.x & 31;
    const int wid  = blockIdx.x * 8 + (threadIdx.x >> 5);
    const int job      = wid >> 10;           // 0..3
    const int rem      = wid & 1023;
    const int row_tile = rem >> 2;            // 0..255
    const int split    = rem & 3;             // 0..3

    const int xbase = (job == 1 || job == 3) ? NROWS : 0;
    const int ybase = (job == 1 || job == 2) ? NROWS : 0;
    const int m0 = row_tile * 16;
    const int hl  = lane >> 4;                // half of wave
    const int l15 = lane & 15;

    // A-fragment source row (ISA: lanes 0-15 and 16-31 both map M=0-15)
    const __hip_bfloat16* xrow_hi = Hi + (size_t)(xbase + m0 + l15) * KDIM;
    const __hip_bfloat16* xrow_lo = Lo + (size_t)(xbase + m0 + l15) * KDIM;

    // Preload norms of my 8 output rows (m = m0 + hl*8 + g)
    float nx[8];
    #pragma unroll
    for (int g = 0; g < 8; ++g) nx[g] = norms[xbase + m0 + hl * 8 + g];

    float rs[8];
    #pragma unroll
    for (int g = 0; g < 8; ++g) rs[g] = 0.0f;

    // Sweep 64 N-tiles (1024 columns), 4 tiles in flight
    for (int blk = 0; blk < 16; ++blk) {
        const int ntile0 = split * 64 + blk * 4;

        v8f acc[4];
        #pragma unroll
        for (int t = 0; t < 4; ++t)
            #pragma unroll
            for (int g = 0; g < 8; ++g) acc[t][g] = 0.0f;

        #pragma unroll
        for (int kc = 0; kc < 4; ++kc) {
            // A fragment: lane<16 holds K[kc*32+0..7] & K[kc*32+16..23];
            //             lane>=16 holds K[+8..15] & K[+24..31]
            const int ka = kc * 32 + hl * 8;
            FragU ahi, alo;
            ahi.u[0] = *(const uint4*)(xrow_hi + ka);
            ahi.u[1] = *(const uint4*)(xrow_hi + ka + 16);
            alo.u[0] = *(const uint4*)(xrow_lo + ka);
            alo.u[1] = *(const uint4*)(xrow_lo + ka + 16);

            #pragma unroll
            for (int t = 0; t < 4; ++t) {
                const int n0 = (ntile0 + t) * 16;
                // B fragment: lane<16 holds K[kc*32+0..7] & K[kc*32+16..23] of col n0+l15;
                //             lane>=16 holds K[+8..15] & K[+24..31]
                const __hip_bfloat16* yh =
                    Hi + (size_t)(ybase + n0 + l15) * KDIM + kc * 32 + hl * 8;
                const __hip_bfloat16* yl =
                    Lo + (size_t)(ybase + n0 + l15) * KDIM + kc * 32 + hl * 8;
                FragU bhi, blo;
                bhi.u[0] = *(const uint4*)(yh);
                bhi.u[1] = *(const uint4*)(yh + 16);
                blo.u[0] = *(const uint4*)(yl);
                blo.u[1] = *(const uint4*)(yl + 16);

                // bf16x3: hi*hi + hi*lo + lo*hi  (~fp32 accuracy)
                acc[t] = wmma_bf16(ahi.v, bhi.v, acc[t]);
                acc[t] = wmma_bf16(ahi.v, blo.v, acc[t]);
                acc[t] = wmma_bf16(alo.v, bhi.v, acc[t]);
            }
        }

        // Epilogue: Cauchy transform + masking + accumulation
        #pragma unroll
        for (int t = 0; t < 4; ++t) {
            const int n0 = (ntile0 + t) * 16;
            const float ny = norms[ybase + n0 + l15];
            #pragma unroll
            for (int g = 0; g < 8; ++g) {
                const int m = m0 + hl * 8 + g;   // local row index (0..4095)
                const int n = n0 + l15;          // local col index (0..4095)
                float dot = acc[t][g];
                float sq  = fmaxf(nx[g] + ny - 2.0f * dot, 0.0f);
                float sim = 1.0f / (sq + 1.0f);
                const bool diag = (m == n);
                if (job < 2 && diag) sim = 0.0f;
                rs[g] += sim;
                if (job == 2 && diag) diaglog[m] = __logf(sim);
            }
        }
    }

    // Butterfly reduction across the 16 lanes of each half (fixed order)
    #pragma unroll
    for (int g = 0; g < 8; ++g) {
        #pragma unroll
        for (int msk = 8; msk >= 1; msk >>= 1)
            rs[g] += __shfl_xor(rs[g], msk, 16);
    }
    if (l15 == 0) {
        const int base = (job * 4 + split) * NROWS + m0 + hl * 8;
        #pragma unroll
        for (int g = 0; g < 8; ++g) partial[base + g] = rs[g];
    }
}

// ---------------------------------------------------------------------------
// Kernel 3: deterministic final reduction -> scalar loss
// ---------------------------------------------------------------------------
__global__ __launch_bounds__(256)
void finalize_kernel(const float* __restrict__ partial,
                     const float* __restrict__ diaglog,
                     float* __restrict__ out) {
    __shared__ float s0[256], s1[256], s2[256];
    const int tid = threadIdx.x;
    float t_align = 0.0f, t1 = 0.0f, t2 = 0.0f;
    for (int i = tid; i < NROWS; i += 256) {
        float saa = 0.0f, sbb = 0.0f, rab = 0.0f, cab = 0.0f;
        #pragma unroll
        for (int s = 0; s < 4; ++s) {
            saa += partial[(0 * 4 + s) * NROWS + i];
            sbb += partial[(1 * 4 + s) * NROWS + i];
            rab += partial[(2 * 4 + s) * NROWS + i];
            cab += partial[(3 * 4 + s) * NROWS + i];
        }
        t_align += diaglog[i];
        t1 += __logf(cab + sbb);   // sum(ab,axis=0)[j] + sum(bb,axis=1)[j]
        t2 += __logf(saa + rab);   // sum(aa,axis=1)[i] + sum(ab,axis=1)[i]
    }
    s0[tid] = t_align; s1[tid] = t1; s2[tid] = t2;
    __syncthreads();
    for (int st = 128; st > 0; st >>= 1) {
        if (tid < st) { s0[tid] += s0[tid + st]; s1[tid] += s1[tid + st]; s2[tid] += s2[tid + st]; }
        __syncthreads();
    }
    if (tid == 0) {
        const float inv = 1.0f / (float)NROWS;
        float align = s0[0] * inv;
        float ls1 = s1[0] * inv;
        float ls2 = s2[0] * inv;
        out[0] = -(align - 0.5f * (ls1 + ls2));
    }
}

// ---------------------------------------------------------------------------
extern "C" void kernel_launch(void* const* d_in, const int* in_sizes, int n_in,
                              void* d_out, int out_size, void* d_ws, size_t ws_size,
                              hipStream_t stream) {
    (void)in_sizes; (void)n_in; (void)out_size; (void)ws_size;
    const float* F = (const float*)d_in[0];          // 8192 x 128 fp32

    char* ws = (char*)d_ws;
    const size_t nelem = (size_t)2 * NROWS * KDIM;   // 8192*128
    __hip_bfloat16* Hi = (__hip_bfloat16*)ws;                       // 2 MB
    __hip_bfloat16* Lo = (__hip_bfloat16*)(ws + nelem * 2);         // 2 MB
    float* norms   = (float*)(ws + nelem * 4);                      // 32 KB
    float* partial = norms + 2 * NROWS;                             // 16*4096 f32 = 256 KB
    float* diaglog = partial + 16 * NROWS;                          // 16 KB

    cvt_norm_kernel<<<2 * NROWS, 128, 0, stream>>>(F, Hi, Lo, norms);
    gram_kernel<<<512, 256, 0, stream>>>(Hi, Lo, norms, partial, diaglog);
    finalize_kernel<<<1, 256, 0, stream>>>(partial, diaglog, (float*)d_out);
}
